// GUNET_15247133901689
// MI455X (gfx1250) — compile-verified, matrix operand order checked by reading, not run on hardware
//
#include <hip/hip_runtime.h>
#include <hip/hip_bf16.h>
#include <math.h>

#define NN      50000
#define HIDF    128
#define EE      800000
#define HSZ     (1u << 21)
#define HMASK   (HSZ - 1)
#define HEMPTY  0xFFFFFFFFFFFFFFFFull
#define CC_ITERS 24
#define DU(a, b) ((unsigned)((((long)(a)) + (b) - 1) / (b)))

typedef float    v2f  __attribute__((ext_vector_type(2)));
typedef float    v8f  __attribute__((ext_vector_type(8)));
typedef _Float16 v16h __attribute__((ext_vector_type(16)));

#if __has_builtin(__builtin_amdgcn_wmma_f32_16x16x4_f32)
#define WMMA_F32_OK 1
#endif

// ---------------- utility kernels ----------------
__global__ void k_zero_f32(float* __restrict__ p, long n) {
  long t = (long)blockIdx.x * blockDim.x + threadIdx.x;
  if (t < n) p[t] = 0.f;
}
__global__ void k_fill_u64(unsigned long long* __restrict__ p, unsigned long long v, long n) {
  long t = (long)blockIdx.x * blockDim.x + threadIdx.x;
  if (t < n) p[t] = v;
}
__global__ void k_iota(int* __restrict__ p, int n) {
  int t = blockIdx.x * blockDim.x + threadIdx.x;
  if (t < n) p[t] = t;
}
__global__ void k_split_edges(const int* __restrict__ ei, int* __restrict__ s,
                              int* __restrict__ d, float* __restrict__ w, int E) {
  int e = blockIdx.x * blockDim.x + threadIdx.x;
  if (e < E) { s[e] = ei[2 * e]; d[e] = ei[2 * e + 1]; w[e] = 1.f; }
}
// dst[i*dstStride + dstOff + f] = src[i*srcStride + f], f in [0,128)
__global__ void k_copy128(const float* __restrict__ src, float* __restrict__ dst,
                          int n, int srcStride, int dstStride, int dstOff) {
  long t = (long)blockIdx.x * blockDim.x + threadIdx.x;
  if (t >= (long)n * HIDF) return;
  int i = (int)(t >> 7), f = (int)(t & 127);
  dst[(long)i * dstStride + dstOff + f] = src[(long)i * srcStride + f];
}
__global__ void k_gather128(const float* __restrict__ x, const int* __restrict__ idx,
                            float* __restrict__ out, int n) {
  long t = (long)blockIdx.x * blockDim.x + threadIdx.x;
  if (t >= (long)n * HIDF) return;
  int i = (int)(t >> 7), f = (int)(t & 127);
  out[t] = x[(long)idx[i] * HIDF + f];
}

// ---------------- GCN conv pieces ----------------
__global__ void k_deg(const int* __restrict__ dst, const float* __restrict__ ew,
                      float* __restrict__ deg, int E) {
  int e = blockIdx.x * blockDim.x + threadIdx.x;
  if (e < E) { float w = ew[e]; if (w != 0.f) atomicAdd(&deg[dst[e]], w); }
}
__global__ void k_dinv(const float* __restrict__ deg, float* __restrict__ dinv, int n) {
  int i = blockIdx.x * blockDim.x + threadIdx.x;
  if (i < n) dinv[i] = rsqrtf(2.f + deg[i]);
}

// C(N x 128) = A(N x K) * B(K x 128); one wave per 16x16 tile, 8 waves/block.
__global__ void __launch_bounds__(256) k_gemm_wmma(const float* __restrict__ A,
                                                   const float* __restrict__ B,
                                                   float* __restrict__ C, int K) {
  int wv = threadIdx.x >> 5, lane = threadIdx.x & 31;
  int hl = lane >> 4, l16 = lane & 15;
  int row0 = blockIdx.x << 4, col0 = wv << 4;
  v8f c = {0.f, 0.f, 0.f, 0.f, 0.f, 0.f, 0.f, 0.f};
  const float* arow = A + (long)(row0 + l16) * K;
#if defined(WMMA_F32_OK)
  for (int k0 = 0; k0 < K; k0 += 4) {
    int ka = k0 + (hl << 1);           // 16x4 f32 A layout: halves hold K{0,1}/{2,3}
    v2f av, bv;
    av.x = arow[ka];
    av.y = arow[ka + 1];
    bv.x = B[(long)ka * HIDF + col0 + l16];
    bv.y = B[(long)(ka + 1) * HIDF + col0 + l16];
    c = __builtin_amdgcn_wmma_f32_16x16x4_f32(false, av, false, bv, (short)0, c,
                                              false, false);
  }
#else
  for (int k0 = 0; k0 < K; k0 += 32) {
    v16h av, bv;
#pragma unroll
    for (int j = 0; j < 8; ++j) {
      int kb = k0 + ((j >= 4) ? 16 : 0) + (hl << 3) + ((j & 3) << 1);
      av[2 * j]     = (_Float16)arow[kb];
      av[2 * j + 1] = (_Float16)arow[kb + 1];
      bv[2 * j]     = (_Float16)B[(long)kb * HIDF + col0 + l16];
      bv[2 * j + 1] = (_Float16)B[(long)(kb + 1) * HIDF + col0 + l16];
    }
    c = __builtin_amdgcn_wmma_f32_16x16x32_f16(false, av, false, bv, (short)0, c,
                                               false, false);
  }
#endif
  // C/D layout: VGPR v -> row = row0 + 8*half + v, col = col0 + l16
  float* crow = C + (long)(row0 + (hl << 3)) * HIDF + col0 + l16;
#pragma unroll
  for (int v = 0; v < 8; ++v) crow[(long)v * HIDF] = c[v];
}

// acc[dst] += xw[src] * (ew * dinv[src] * dinv[dst]); 32 threads/edge, float4 each
__global__ void k_scatter128(const int* __restrict__ src, const int* __restrict__ dst,
                             const float* __restrict__ ew, const float* __restrict__ dinv,
                             const float* __restrict__ xw, float* __restrict__ acc, int E) {
  long t = (long)blockIdx.x * blockDim.x + threadIdx.x;
  int e = (int)(t >> 5);
  if (e >= E) return;
  float w = ew[e];
  if (w == 0.f) return;
  int s = src[e], d = dst[e];
  float nrm = w * dinv[s] * dinv[d];
  int q = ((int)t & 31) << 2;
  float4 v = *(const float4*)(xw + (long)s * HIDF + q);
  float* o = acc + (long)d * HIDF + q;
  atomicAdd(o + 0, v.x * nrm);
  atomicAdd(o + 1, v.y * nrm);
  atomicAdd(o + 2, v.z * nrm);
  atomicAdd(o + 3, v.w * nrm);
}
__global__ void k_finalize128(const float* __restrict__ acc, const float* __restrict__ xw,
                              const float* __restrict__ dinv, const float* __restrict__ bias,
                              float* __restrict__ out, int relu, int n) {
  long t = (long)blockIdx.x * blockDim.x + threadIdx.x;
  if (t >= (long)n * HIDF) return;
  int i = (int)(t >> 7), f = (int)(t & 127);
  float di = dinv[i];
  float v = acc[t] + 2.f * di * di * xw[t] + bias[f];
  out[t] = relu ? fmaxf(v, 0.f) : v;
}

// ---------------- ClusterPooling ----------------
__global__ void k_edge_score(const float* __restrict__ h, const int* __restrict__ src,
                             const int* __restrict__ dst, const float* __restrict__ ew,
                             const float* __restrict__ Wp, const float* __restrict__ bp,
                             float* __restrict__ score, int* __restrict__ sel, int E) {
  long t = (long)blockIdx.x * blockDim.x + threadIdx.x;
  int e = (int)(t >> 5), lane = (int)(t & 31);
  if (e >= E) return;
  int s = src[e], d = dst[e];
  const float* hs = h + (long)s * HIDF;
  const float* hd = h + (long)d * HIDF;
  float a = 0.f;
#pragma unroll
  for (int f = lane; f < HIDF; f += 32) a += hs[f] * Wp[f] + hd[f] * Wp[HIDF + f];
#pragma unroll
  for (int off = 16; off > 0; off >>= 1) a += __shfl_xor(a, off, 32);
  if (lane == 0) {
    float sc = 1.f / (1.f + __expf(-(a + bp[0])));
    score[e] = sc;
    sel[e] = (ew[e] > 0.f && s != d && sc > 0.5f) ? 1 : 0;
  }
}
__global__ void k_cc_edge(const int* __restrict__ src, const int* __restrict__ dst,
                          const int* __restrict__ sel, int* __restrict__ lab, int E) {
  int e = blockIdx.x * blockDim.x + threadIdx.x;
  if (e >= E || !sel[e]) return;
  int a = lab[src[e]], b = lab[dst[e]];
  int m = a < b ? a : b;
  atomicMin(&lab[src[e]], m);
  atomicMin(&lab[dst[e]], m);
}
__global__ void k_cc_jump(int* __restrict__ lab, int n) {
  int i = blockIdx.x * blockDim.x + threadIdx.x;
  if (i >= n) return;
  int l = lab[i];
  int l2 = lab[l];
  if (l2 < l) lab[i] = l2;
}
__global__ void k_pool_stats(const int* __restrict__ src, const int* __restrict__ sel,
                             const float* __restrict__ score, const int* __restrict__ lab,
                             float* __restrict__ ssum, float* __restrict__ scnt, int E) {
  int e = blockIdx.x * blockDim.x + threadIdx.x;
  if (e >= E || !sel[e]) return;
  int c = lab[src[e]];
  atomicAdd(&ssum[c], score[e]);
  atomicAdd(&scnt[c], 1.f);
}
__global__ void k_pool_w(const float* __restrict__ ssum, const float* __restrict__ scnt,
                         float* __restrict__ w, int n) {
  int i = blockIdx.x * blockDim.x + threadIdx.x;
  if (i >= n) return;
  float c = scnt[i];
  w[i] = (c > 0.f) ? (ssum[i] / fmaxf(c, 1.f)) : 1.f;
}
__global__ void k_pool_newx(const float* __restrict__ h, const int* __restrict__ lab,
                            float* __restrict__ newx, int n) {
  long t = (long)blockIdx.x * blockDim.x + threadIdx.x;
  int i = (int)(t >> 5);
  if (i >= n) return;
  int q = ((int)t & 31) << 2;
  int c = lab[i];
  float4 v = *(const float4*)(h + (long)i * HIDF + q);
  float* o = newx + (long)c * HIDF + q;
  atomicAdd(o + 0, v.x);
  atomicAdd(o + 1, v.y);
  atomicAdd(o + 2, v.z);
  atomicAdd(o + 3, v.w);
}
__global__ void k_scale_rows(float* __restrict__ x, const float* __restrict__ w, int n) {
  long t = (long)blockIdx.x * blockDim.x + threadIdx.x;
  if (t >= (long)n * HIDF) return;
  x[t] *= w[t >> 7];
}
// contracted-edge dedup via hash set (set-equivalent to reference lexsort dedup)
__global__ void k_edge_contract(const int* __restrict__ src, const int* __restrict__ dst,
                                const float* __restrict__ ew, const int* __restrict__ lab,
                                unsigned long long* __restrict__ tab, int* __restrict__ nsrc,
                                int* __restrict__ ndst, float* __restrict__ new_ew, int E) {
  int e = blockIdx.x * blockDim.x + threadIdx.x;
  if (e >= E) return;
  int a = 0, b = 0;
  float w = 0.f;
  if (ew[e] > 0.f) {
    int ca = lab[src[e]], cb = lab[dst[e]];
    if (ca != cb) {
      unsigned long long key = (unsigned long long)ca * 50000ull + (unsigned long long)cb;
      unsigned long long slot = (key * 0x9E3779B97F4A7C15ull >> 40) & HMASK;
      for (;;) {
        unsigned long long prev = atomicCAS(&tab[slot], HEMPTY, key);
        if (prev == HEMPTY) { a = ca; b = cb; w = 1.f; break; }  // first occurrence
        if (prev == key) break;                                   // duplicate -> drop
        slot = (slot + 1) & HMASK;
      }
    }
  }
  nsrc[e] = a; ndst[e] = b; new_ew[e] = w;
}

// ---------------- final 384 -> 1 layer ----------------
__global__ void k_gemv(const float* __restrict__ A, const float* __restrict__ W,
                       float* __restrict__ out, int n, int K) {
  long t = (long)blockIdx.x * blockDim.x + threadIdx.x;
  int i = (int)(t >> 5), lane = (int)(t & 31);
  if (i >= n) return;
  const float* a = A + (long)i * K;
  float acc = 0.f;
  for (int f = lane; f < K; f += 32) acc += a[f] * W[f];
#pragma unroll
  for (int off = 16; off > 0; off >>= 1) acc += __shfl_xor(acc, off, 32);
  if (lane == 0) out[i] = acc;
}
__global__ void k_scatter1(const int* __restrict__ src, const int* __restrict__ dst,
                           const float* __restrict__ ew, const float* __restrict__ dinv,
                           const float* __restrict__ xw, float* __restrict__ acc, int E) {
  int e = blockIdx.x * blockDim.x + threadIdx.x;
  if (e >= E) return;
  float w = ew[e];
  if (w == 0.f) return;
  int s = src[e], d = dst[e];
  atomicAdd(&acc[d], xw[s] * w * dinv[s] * dinv[d]);
}
__global__ void k_final_sig(const float* __restrict__ acc, const float* __restrict__ xw,
                            const float* __restrict__ dinv, const float* __restrict__ bias,
                            float* __restrict__ out, int n) {
  int i = blockIdx.x * blockDim.x + threadIdx.x;
  if (i >= n) return;
  float di = dinv[i];
  float v = acc[i] + 2.f * di * di * xw[i] + bias[0];
  out[i] = 1.f / (1.f + __expf(-v));
}

// ---------------- orchestration ----------------
extern "C" void kernel_launch(void* const* d_in, const int* in_sizes, int n_in,
                              void* d_out, int out_size, void* d_ws, size_t ws_size,
                              hipStream_t stream) {
  (void)in_sizes; (void)n_in; (void)out_size; (void)ws_size;
  const float* x_in = (const float*)d_in[0];
  const int*   ei   = (const int*)d_in[1];
  const float* Wd[4] = {(const float*)d_in[3], (const float*)d_in[5],
                        (const float*)d_in[7], (const float*)d_in[9]};
  const float* bd[4] = {(const float*)d_in[4], (const float*)d_in[6],
                        (const float*)d_in[8], (const float*)d_in[10]};
  const float* Wp[3] = {(const float*)d_in[11], (const float*)d_in[13], (const float*)d_in[15]};
  const float* bp[3] = {(const float*)d_in[12], (const float*)d_in[14], (const float*)d_in[16]};
  const float* Wu[3] = {(const float*)d_in[17], (const float*)d_in[19], (const float*)d_in[21]};
  const float* bu[3] = {(const float*)d_in[18], (const float*)d_in[20], (const float*)d_in[22]};

  // workspace carve-out
  char* base = (char*)d_ws;
  size_t off = 0;
  auto carve = [&](size_t bytes) -> void* {
    void* p = base + off;
    off += (bytes + 255) & ~(size_t)255;
    return p;
  };
  int* srcL[4]; int* dstL[4]; float* ewL[4];
  for (int i = 0; i < 4; ++i) {
    srcL[i] = (int*)carve((size_t)EE * 4);
    dstL[i] = (int*)carve((size_t)EE * 4);
    ewL[i]  = (float*)carve((size_t)EE * 4);
  }
  int* clu[3];
  for (int i = 0; i < 3; ++i) clu[i] = (int*)carve((size_t)NN * 4);
  float* mem[3];
  for (int i = 0; i < 3; ++i) mem[i] = (float*)carve((size_t)NN * HIDF * 4);
  float* xw  = (float*)carve((size_t)NN * HIDF * 4);
  float* acc = (float*)carve((size_t)NN * HIDF * 4);
  float* xA  = (float*)carve((size_t)NN * HIDF * 4);
  float* xB  = (float*)carve((size_t)NN * HIDF * 4);
  float* cat = (float*)carve((size_t)NN * 384 * 4);
  float* deg  = (float*)carve((size_t)NN * 4);
  float* dinv = (float*)carve((size_t)NN * 4);
  float* ssum = (float*)carve((size_t)NN * 4);
  float* scnt = (float*)carve((size_t)NN * 4);
  float* wcl  = (float*)carve((size_t)NN * 4);
  float* score = (float*)carve((size_t)EE * 4);
  int*   sel   = (int*)carve((size_t)EE * 4);
  unsigned long long* htab = (unsigned long long*)carve((size_t)HSZ * 8);

  auto Z = [&](float* p, long n) {
    k_zero_f32<<<DU(n, 256), 256, 0, stream>>>(p, n);
  };
  // full GCNConv(improved) with M=128: out = segsum + 2*dinv^2*xw + b, opt relu
  auto gcn128 = [&](const float* x, int K, const int* s, const int* d, const float* w,
                    const float* W, const float* b, float* out, int relu) {
    Z(deg, NN);
    Z(acc, (long)NN * HIDF);
    k_deg<<<DU(EE, 256), 256, 0, stream>>>(d, w, deg, EE);
    k_dinv<<<DU(NN, 256), 256, 0, stream>>>(deg, dinv, NN);
    k_gemm_wmma<<<NN / 16, 256, 0, stream>>>(x, W, xw, K);
    k_scatter128<<<DU((long)EE * 32, 256), 256, 0, stream>>>(s, d, w, dinv, xw, acc, EE);
    k_finalize128<<<DU((long)NN * HIDF, 256), 256, 0, stream>>>(acc, xw, dinv, b, out, relu, NN);
  };

  // level-0 edges
  k_split_edges<<<DU(EE, 256), 256, 0, stream>>>(ei, srcL[0], dstL[0], ewL[0], EE);

  // ---- encoder (down) ----
  const float* x = x_in;
  for (int i = 0; i < 3; ++i) {
    gcn128(x, HIDF, srcL[i], dstL[i], ewL[i], Wd[i], bd[i], mem[i], 1);
    // cluster pooling on h = mem[i]
    k_edge_score<<<DU((long)EE * 32, 256), 256, 0, stream>>>(mem[i], srcL[i], dstL[i], ewL[i],
                                                             Wp[i], bp[i], score, sel, EE);
    k_iota<<<DU(NN, 256), 256, 0, stream>>>(clu[i], NN);
    for (int it = 0; it < CC_ITERS; ++it) {
      k_cc_edge<<<DU(EE, 256), 256, 0, stream>>>(srcL[i], dstL[i], sel, clu[i], EE);
      k_cc_jump<<<DU(NN, 256), 256, 0, stream>>>(clu[i], NN);
      k_cc_jump<<<DU(NN, 256), 256, 0, stream>>>(clu[i], NN);
    }
    Z(ssum, NN);
    Z(scnt, NN);
    k_pool_stats<<<DU(EE, 256), 256, 0, stream>>>(srcL[i], sel, score, clu[i], ssum, scnt, EE);
    k_pool_w<<<DU(NN, 256), 256, 0, stream>>>(ssum, scnt, wcl, NN);
    Z(xA, (long)NN * HIDF);
    k_pool_newx<<<DU((long)NN * 32, 256), 256, 0, stream>>>(mem[i], clu[i], xA, NN);
    k_scale_rows<<<DU((long)NN * HIDF, 256), 256, 0, stream>>>(xA, wcl, NN);
    k_fill_u64<<<DU(HSZ, 256), 256, 0, stream>>>(htab, HEMPTY, HSZ);
    k_edge_contract<<<DU(EE, 256), 256, 0, stream>>>(srcL[i], dstL[i], ewL[i], clu[i], htab,
                                                     srcL[i + 1], dstL[i + 1], ewL[i + 1], EE);
    x = xA;
  }

  // ---- bottleneck ----
  gcn128(xA, HIDF, srcL[3], dstL[3], ewL[3], Wd[3], bd[3], xB, 0);

  // ---- decoder (up), levels 2 and 1: K = 256 ----
  for (int j = 0; j < 2; ++j) {
    int lv = 2 - j;
    k_gather128<<<DU((long)NN * HIDF, 256), 256, 0, stream>>>(xB, clu[lv], xA, NN);
    k_copy128<<<DU((long)NN * HIDF, 256), 256, 0, stream>>>(mem[lv], cat, NN, HIDF, 256, 0);
    k_copy128<<<DU((long)NN * HIDF, 256), 256, 0, stream>>>(xA, cat, NN, HIDF, 256, HIDF);
    gcn128(cat, 256, srcL[lv], dstL[lv], ewL[lv], Wu[j], bu[j], xB, 1);
  }

  // ---- final level 0: cat384 = [h0 | x_in | x[cluster0]], Wu2: 384 -> 1 ----
  k_gather128<<<DU((long)NN * HIDF, 256), 256, 0, stream>>>(xB, clu[0], xA, NN);
  k_copy128<<<DU((long)NN * HIDF, 256), 256, 0, stream>>>(mem[0], cat, NN, HIDF, 384, 0);
  k_copy128<<<DU((long)NN * HIDF, 256), 256, 0, stream>>>(x_in, cat, NN, HIDF, 384, HIDF);
  k_copy128<<<DU((long)NN * HIDF, 256), 256, 0, stream>>>(xA, cat, NN, HIDF, 384, 256);
  Z(deg, NN);
  k_deg<<<DU(EE, 256), 256, 0, stream>>>(dstL[0], ewL[0], deg, EE);
  k_dinv<<<DU(NN, 256), 256, 0, stream>>>(deg, dinv, NN);
  k_gemv<<<DU((long)NN * 32, 256), 256, 0, stream>>>(cat, Wu[2], xw, NN, 384);
  Z(acc, NN);
  k_scatter1<<<DU(EE, 256), 256, 0, stream>>>(srcL[0], dstL[0], ewL[0], dinv, xw, acc, EE);
  k_final_sig<<<DU(NN, 256), 256, 0, stream>>>(acc, xw, dinv, bu[2], (float*)d_out, NN);
}
